// Performer_56100862820683
// MI455X (gfx1250) — matrix operand order repaired and checked by reading, not compiled, not run on hardware
//
#include <hip/hip_runtime.h>
#include <hip/hip_bf16.h>

// Performer (FAVOR+) attention for B=4,H=16,S=8192,D=64,M=64 on gfx1250.
// Memory-bound (~0.54 GB traffic vs ~17 GFLOP): stream q/k/v once, do all
// GEMMs with v_wmma_f32_16x16x32_f16 (f16 in, f32 accum), keep per-head
// buf1 (64x80 f32) in a 1.25 MB workspace (L2-resident on 192MB L2).
// All WMMA operands live in LDS in "operand-major" layout (A: [row][K],
// B: [col][K]) so every fragment load is 2x ds_load_b128 per lane.

typedef __attribute__((ext_vector_type(16))) _Float16 v16h;
typedef __attribute__((ext_vector_type(8)))  _Float16 v8h;
typedef __attribute__((ext_vector_type(8)))  float    v8f;

#define S_TOT   8192
#define D_DIM   64
#define M_DIM   64
#define NPAD    80            // 65 cols (V||1) padded to 5 n-tiles of 16
#define TILE_S  32
#define CHUNKS  16            // phase-1 S-chunks per head
#define ROWS_PER_CHUNK (S_TOT / CHUNKS)   // 512
#define NHEADS  64            // B*H
#define LD64    72            // halves; 144B rows: 16B-aligned, conflict-free
#define LD32    40            // halves; 80B rows: 16B-aligned, conflict-free

union F8  { v8f  v; float    f[8];  };
union H8  { v8h  v; _Float16 h[8];  };
union H16 { v16h v; v8h      p[2];  };

__device__ __forceinline__ v8f wmma_f16(v16h a, v16h b, v8f c) {
    return __builtin_amdgcn_wmma_f32_16x16x32_f16(
        false, a, false, b, (short)0, c, false, false);
}

// A fragment (16 rows x K=32) from row-major [row][K] storage.
// lane<16: e0..7 = K kb+0..7, e8..15 = K kb+16..23 ; lane>=16: +8 on both.
__device__ __forceinline__ v16h load_a_rm(const _Float16* src, int ld,
                                          int rb, int kb, int lh, int lm) {
    const _Float16* p = src + (rb + lm) * ld + kb + (lh << 3);
    H16 u;
    u.p[0] = *(const v8h*)p;
    u.p[1] = *(const v8h*)(p + 16);
    return u.v;
}

// B fragment (K=32 x 16 cols) from transposed [col][K] storage.
// lane<16: e0..7 = K kb+0..7, e8..15 = K kb+16..23 ; lane>=16: +8 on both (col = lane%16).
__device__ __forceinline__ v16h load_b_tr(const _Float16* srcT, int ld,
                                          int kb, int nb, int lh, int lm) {
    const _Float16* p = srcT + (nb + lm) * ld + kb + (lh << 3);
    H16 u;
    u.p[0] = *(const v8h*)p;
    u.p[1] = *(const v8h*)(p + 16);
    return u.v;
}

__device__ __forceinline__ v8h pack8(float4 a, float4 b) {
    H8 u;
    u.h[0] = (_Float16)a.x; u.h[1] = (_Float16)a.y;
    u.h[2] = (_Float16)a.z; u.h[3] = (_Float16)a.w;
    u.h[4] = (_Float16)b.x; u.h[5] = (_Float16)b.y;
    u.h[6] = (_Float16)b.z; u.h[7] = (_Float16)b.w;
    return u.v;
}

__device__ __forceinline__ float dot8(float4 a, float4 b) {
    float s = a.x * a.x;
    s = fmaf(a.y, a.y, s); s = fmaf(a.z, a.z, s); s = fmaf(a.w, a.w, s);
    s = fmaf(b.x, b.x, s); s = fmaf(b.y, b.y, s);
    s = fmaf(b.z, b.z, s); s = fmaf(b.w, b.w, s);
    return s;
}

__global__ void __launch_bounds__(256) performer_zero_buf1(float* buf1) {
    int i = blockIdx.x * 256 + threadIdx.x;
    if (i < NHEADS * M_DIM * NPAD) buf1[i] = 0.0f;
}

// ---------------- Phase 1: buf1[head] += k'^T @ [V||1] ----------------
__global__ void __launch_bounds__(256) performer_phase1(
    const float* __restrict__ key, const float* __restrict__ value,
    const float* __restrict__ omega, float* __restrict__ buf1) {

    __shared__ __align__(16) _Float16 ogT[M_DIM][LD64];  // omega^T: [m][d]
    __shared__ __align__(16) _Float16 ks [TILE_S][LD64]; // key tile: [s][d]
    __shared__ __align__(16) _Float16 kpT[M_DIM][LD32];  // k'^T: [m][s]
    __shared__ __align__(16) _Float16 ctT[NPAD][LD32];   // [V||1||0]^T: [c][s]
    __shared__ float hraw[TILE_S];   // atomic sum-of-squares accumulator
    __shared__ float hrow[TILE_S];   // final h(x) per row

    const int tid  = threadIdx.x;
    const int lane = tid & 31;
    const int w    = tid >> 5;
    const int head = blockIdx.y;
    const int chunk = blockIdx.x;
    const int lh = lane >> 4, lm = lane & 15;

    const float* kh = key   + (size_t)head * S_TOT * D_DIM;
    const float* vh = value + (size_t)head * S_TOT * D_DIM;
    float*       b1 = buf1  + (size_t)head * M_DIM * NPAD;

    // stage omega transposed once: ogT[m][d] = omega[d*64+m]
    for (int i = tid; i < D_DIM * M_DIM; i += 256)
        ogT[i & 63][i >> 6] = (_Float16)omega[i];
    // constant rows of ctT: ones row (c=64) and zero pad rows (c=65..79)
    if (tid < TILE_S) { ctT[64][tid] = (_Float16)1.0f; hraw[tid] = 0.0f; }
    for (int i = tid; i < 15 * TILE_S; i += 256)
        ctT[65 + (i >> 5)][i & 31] = (_Float16)0.0f;

    // persistent per-wave buf1 accumulator tiles (16x16 each)
    F8 acc0, acc1, acc2;
#pragma unroll
    for (int j = 0; j < 8; ++j) { acc0.f[j] = 0.f; acc1.f[j] = 0.f; acc2.f[j] = 0.f; }
    const int t0 = w, t1 = w + 8, t2 = (w < 4) ? (w + 16) : -1;

    __syncthreads();

    for (int it = 0; it < ROWS_PER_CHUNK / TILE_S; ++it) {
        const int s_base = chunk * ROWS_PER_CHUNK + it * TILE_S;

        // stage 32 rows of key (row-major f16) and value (transposed f16);
        // fold the per-row |k|^2 partial into an LDS f32 atomic (ds_add_f32)
        {
            const float4* k4 = (const float4*)(kh + (size_t)s_base * D_DIM);
            const float4* v4 = (const float4*)(vh + (size_t)s_base * D_DIM);
            const int r  = tid >> 3;        // 0..31
            const int c0 = (tid & 7) << 3;  // 0,8,...,56
            float4 ka = k4[tid * 2], kb2 = k4[tid * 2 + 1];
            *(v8h*)&ks[r][c0] = pack8(ka, kb2);
            atomicAdd(&hraw[r], dot8(ka, kb2));
            float4 va = v4[tid * 2], vb = v4[tid * 2 + 1];
            H8 u; u.v = pack8(va, vb);
#pragma unroll
            for (int e = 0; e < 8; ++e) ctT[c0 + e][r] = u.h[e];
        }
        __syncthreads();

        if (tid < TILE_S)
            hrow[tid] = __expf(-0.5f * hraw[tid]) * 0.125f;   // 1/sqrt(M)=0.125
        __syncthreads();

        // k' = h * exp(k @ omega): wave w -> row-subtile r=w>>2, n-tile nt=w&3
        {
            const int r    = (w >> 2) << 4;
            const int nt16 = (w & 3) << 4;
            v16h a0 = load_a_rm(&ks[0][0], LD64, r, 0,  lh, lm);
            v16h a1 = load_a_rm(&ks[0][0], LD64, r, 32, lh, lm);
            v16h b0 = load_b_tr(&ogT[0][0], LD64, 0,  nt16, lh, lm);
            v16h b1v = load_b_tr(&ogT[0][0], LD64, 32, nt16, lh, lm);
            F8 c; c.v = (v8f){};
            c.v = wmma_f16(a0, b0, c.v);
            c.v = wmma_f16(a1, b1v, c.v);
            // store transposed: kpT[m][s], 8 contiguous s per lane -> b128
            H8 pk;
#pragma unroll
            for (int j = 0; j < 8; ++j) {
                int sr = r + (lh << 3) + j;
                pk.h[j] = (_Float16)(hrow[sr] * __expf(c.f[j]));
            }
            *(v8h*)&kpT[nt16 + lm][r + (lh << 3)] = pk.v;
        }
        __syncthreads();

        // buf1 += kp^T (64 x 32) @ ct (32 x 80): 20 tiles across 8 waves
        {
            auto do_tile = [&](int tidx, F8& acc) {
                int mt = (tidx / 5) << 4, nt16 = (tidx % 5) << 4;
                v16h a = load_a_rm(&kpT[0][0], LD32, mt, 0, lh, lm);
                v16h b = load_b_tr(&ctT[0][0], LD32, 0, nt16, lh, lm);
                acc.v = wmma_f16(a, b, acc.v);
            };
            do_tile(t0, acc0);
            do_tile(t1, acc1);
            if (t2 >= 0) do_tile(t2, acc2);
        }
        if (tid < TILE_S) hraw[tid] = 0.0f;   // reset for next iteration
        __syncthreads();
    }

    // flush accumulators (multiple chunks per head -> atomic f32)
    auto flush = [&](int tidx, F8& acc) {
        int mt = (tidx / 5) << 4, nt16 = (tidx % 5) << 4;
#pragma unroll
        for (int j = 0; j < 8; ++j)
            atomicAdd(&b1[(mt + j + (lh << 3)) * NPAD + nt16 + lm], acc.f[j]);
    };
    flush(t0, acc0);
    flush(t1, acc1);
    if (t2 >= 0) flush(t2, acc2);
}

// ------------- Phase 2: out = (q' @ buf1)[:, :64] / [:, 64] -------------
__global__ void __launch_bounds__(256) performer_phase2(
    const float* __restrict__ query, const float* __restrict__ buf1,
    const float* __restrict__ omega, float* __restrict__ out) {

    __shared__ __align__(16) _Float16 ogT[M_DIM][LD64];      // [m][d]
    __shared__ __align__(16) _Float16 bqT[NPAD][LD64];       // buf1^T: [n][m]
    __shared__ __align__(16) _Float16 qs[8][16][LD64];       // per-wave [s][d]/[s][m]
    __shared__ float hq[8][16];
    __shared__ float rden[8][16];    // reciprocal denominators

    const int tid  = threadIdx.x;
    const int lane = tid & 31;
    const int w    = tid >> 5;
    const int head = blockIdx.y;
    const int lh = lane >> 4, lm = lane & 15;

    const float* b1 = buf1 + (size_t)head * M_DIM * NPAD;
    for (int i = tid; i < D_DIM * M_DIM; i += 256)
        ogT[i & 63][i >> 6] = (_Float16)omega[i];
    for (int i = tid; i < M_DIM * NPAD; i += 256)
        bqT[i % NPAD][i / NPAD] = (_Float16)b1[i];
    __syncthreads();

    const int s0 = blockIdx.x * 128 + w * 16;
    const float* qh = query + ((size_t)head * S_TOT + s0) * D_DIM; // 1024 contiguous
    {
        const float4* q4 = (const float4*)qh;
        const int r  = lane >> 1;            // 0..15 (2 lanes per row)
        const int c0 = (lane & 1) << 5;      // 0 or 32
        float ss = 0.0f;
#pragma unroll
        for (int u = 0; u < 4; ++u) {
            float4 a = q4[lane * 8 + u * 2], b = q4[lane * 8 + u * 2 + 1];
            *(v8h*)&qs[w][r][c0 + (u << 3)] = pack8(a, b);
            ss += dot8(a, b);
        }
        ss += __shfl_xor(ss, 1, 32);         // combine the two half-rows
        if ((lane & 1) == 0)
            hq[w][r] = __expf(-0.5f * ss) * 0.125f;
    }

    // q' = h * exp(q @ omega) for this wave's 16 rows
    v16h a0 = load_a_rm(&qs[w][0][0], LD64, 0, 0,  lh, lm);
    v16h a1 = load_a_rm(&qs[w][0][0], LD64, 0, 32, lh, lm);
    F8 c[4];
#pragma unroll
    for (int nt = 0; nt < 4; ++nt) {
        c[nt].v = (v8f){};
        c[nt].v = wmma_f16(a0, load_b_tr(&ogT[0][0], LD64, 0,  nt << 4, lh, lm), c[nt].v);
        c[nt].v = wmma_f16(a1, load_b_tr(&ogT[0][0], LD64, 32, nt << 4, lh, lm), c[nt].v);
    }
#pragma unroll
    for (int nt = 0; nt < 4; ++nt)
#pragma unroll
        for (int j = 0; j < 8; ++j) {
            int r = j + (lh << 3);
            qs[w][r][(nt << 4) + lm] = (_Float16)(hq[w][r] * __expf(c[nt].f[j]));
        }

    // buf2 = q' (16x64) @ buf1 (64x80)
    v16h p0 = load_a_rm(&qs[w][0][0], LD64, 0, 0,  lh, lm);
    v16h p1 = load_a_rm(&qs[w][0][0], LD64, 0, 32, lh, lm);
    F8 o[5];
#pragma unroll
    for (int nt = 0; nt < 5; ++nt) {
        o[nt].v = (v8f){};
        o[nt].v = wmma_f16(p0, load_b_tr(&bqT[0][0], LD64, 0,  nt << 4, lh, lm), o[nt].v);
        o[nt].v = wmma_f16(p1, load_b_tr(&bqT[0][0], LD64, 32, nt << 4, lh, lm), o[nt].v);
    }

    // reciprocal denominator = 1 / column 64 (n-tile 4, lane%16 == 0)
    if (lm == 0) {
#pragma unroll
        for (int j = 0; j < 8; ++j) rden[w][j + (lh << 3)] = 1.0f / o[4].f[j];
    }

    float* oh = out + ((size_t)head * S_TOT + s0) * D_DIM;
#pragma unroll
    for (int nt = 0; nt < 4; ++nt)
#pragma unroll
        for (int j = 0; j < 8; ++j) {
            int r = j + (lh << 3);
            oh[(size_t)r * D_DIM + (nt << 4) + lm] = o[nt].f[j] * rden[w][r];
        }
}

extern "C" void kernel_launch(void* const* d_in, const int* in_sizes, int n_in,
                              void* d_out, int out_size, void* d_ws, size_t ws_size,
                              hipStream_t stream) {
    const float* query = (const float*)d_in[0];
    const float* key   = (const float*)d_in[1];
    const float* value = (const float*)d_in[2];
    const float* omega = (const float*)d_in[3];
    float* out  = (float*)d_out;
    float* buf1 = (float*)d_ws;   // NHEADS * 64 * 80 f32 = 1.25 MB

    const int buf1_elems = NHEADS * M_DIM * NPAD;
    performer_zero_buf1<<<(buf1_elems + 255) / 256, 256, 0, stream>>>(buf1);
    performer_phase1<<<dim3(CHUNKS, NHEADS), 256, 0, stream>>>(key, value, omega, buf1);
    performer_phase2<<<dim3(S_TOT / 128, NHEADS), 256, 0, stream>>>(query, buf1, omega, out);
}
